// HGRN2Attention_43009802502714
// MI455X (gfx1250) — matrix operand order repaired and checked by reading, not compiled, not run on hardware
//
#include <hip/hip_runtime.h>
#include <hip/hip_bf16.h>

// ---------------------------------------------------------------------------
// HGRN2 gated linear attention for MI455X (gfx1250), chunked-GLA formulation.
// Dense math on v_wmma_f32_16x16x32_bf16; A-tile staging via CDNA5 async
// global->LDS DMA (ASYNCcnt) with double-buffered software pipelining.
// ---------------------------------------------------------------------------

#define NB 2
#define NL 2048
#define NHID 2048
#define NNH 16
#define NDK 128
#define NDV 128
#define QSCALE 0.08838834764831845f   // 128^-0.5
#define REPS 1e-5f

typedef __attribute__((ext_vector_type(16))) __bf16 v16bf;
typedef __attribute__((ext_vector_type(8)))  float  v8f;
typedef __attribute__((ext_vector_type(4)))  unsigned int v4u;
typedef __attribute__((ext_vector_type(4)))  float  v4f;

__device__ __forceinline__ unsigned short f2bf(float x) {
  unsigned int u = __builtin_bit_cast(unsigned int, x);
  unsigned int r = u + 0x7FFFu + ((u >> 16) & 1u);       // round-to-nearest-even
  return (unsigned short)(r >> 16);
}
__device__ __forceinline__ float bf2f(unsigned short h) {
  unsigned int u = ((unsigned int)h) << 16;
  return __builtin_bit_cast(float, u);
}

struct Frag { union { v4u u[2]; v16bf bf; }; };

// A-matrix fragment (16x32 bf16): lane m=lane&15, half=lane>>4 holds
// K = half*16 .. half*16+15 contiguous: two adjacent 16B chunks.
__device__ __forceinline__ Frag loadA16(const unsigned short* p) {
  Frag f;
  f.u[0] = *(const v4u*)p;
  f.u[1] = *(const v4u*)(p + 8);
  return f;
}
// B-matrix fragment (32x16 bf16) from a [N][K]-transposed LDS tile: lane n=lane&15
// holds K = half*16 .. half*16+15 contiguous -> 32 contiguous bytes.
__device__ __forceinline__ Frag loadB16(const unsigned short* p) {
  Frag f;
  f.u[0] = *(const v4u*)p;
  f.u[1] = *(const v4u*)(p + 8);
  return f;
}

__device__ __forceinline__ v8f wmma_bf16(v16bf a, v16bf b, v8f c) {
  return __builtin_amdgcn_wmma_f32_16x16x32_bf16(false, a, false, b, (short)0, c,
                                                 false, false);
}

// ---- CDNA5 async global->LDS DMA (per-lane 16B), tracked with ASYNCcnt ----
__device__ __forceinline__ void async_copy_b128(unsigned lds_byte_addr,
                                                const void* gaddr) {
  asm volatile("global_load_async_to_lds_b128 %0, %1, off"
               :: "v"(lds_byte_addr), "v"(gaddr)
               : "memory");
}
__device__ __forceinline__ void wait_asynccnt0() {
  asm volatile("s_wait_asynccnt 0x0" ::: "memory");
}
__device__ __forceinline__ unsigned lds_addr_of(const void* p) {
  return (unsigned)(uintptr_t)p;      // low 32 bits of generic ptr = LDS offset
}

// ---------------------------------------------------------------------------
// f32 -> bf16 conversion
// ---------------------------------------------------------------------------
__global__ __launch_bounds__(256) void cvt_f32_bf16(const float* __restrict__ x,
                                                    unsigned short* __restrict__ y,
                                                    int n) {
  int i = blockIdx.x * 256 + threadIdx.x;
  if (i < n) y[i] = f2bf(x[i]);
}

// ---------------------------------------------------------------------------
// bf16 WMMA GEMM: C[M,N] = A[M,K] @ B[K,N], 128x128 tile per workgroup,
// 8 waves, each a 32x64 sub-tile = 2x4 wmma 16x16 tiles, BK=32,
// double-buffered LDS, async A staging, pipelined B staging.
// Epilogue modes:
//   0: out32[r*N+c] = acc                      (final Wo GEMM, row-major f32)
//   1: out16[bhld]  = bf16(relu(acc)*scale)    (q projection, head layout)
//   2: out16[bhld]  = bf16(sigmoid(acc))       (k)
//      out32[bhld]  = -softplus(acc)           (g = log(1-sigmoid))
//   3: out16[bhld]  = bf16(acc)                (v projection)
// head layout index: ((b*NH+h)*L + l)*128 + d
// ---------------------------------------------------------------------------
#define ABUF (128 * 40)

__global__ __launch_bounds__(256) void gemm_bf16_wmma(
    const unsigned short* __restrict__ A, const unsigned short* __restrict__ Bw,
    int M, int N, int K, int mode, float scale,
    unsigned short* __restrict__ out16, float* __restrict__ out32) {
  __shared__ __attribute__((aligned(16))) unsigned short As[2 * ABUF];
  __shared__ __attribute__((aligned(16))) unsigned short Bt[2 * ABUF];

  const int tid  = threadIdx.x;
  const int lane = tid & 31;
  const int w    = tid >> 5;
  const int m0 = blockIdx.y * 128;
  const int n0 = blockIdx.x * 128;
  const int mBase = (w & 3) * 32;
  const int nBase = (w >> 2) * 64;
  const int ml   = lane & 15;
  const int half = lane >> 4;

  v8f zero = {0.f, 0.f, 0.f, 0.f, 0.f, 0.f, 0.f, 0.f};
  v8f c[2][4];
#pragma unroll
  for (int mt = 0; mt < 2; ++mt)
#pragma unroll
    for (int nt = 0; nt < 4; ++nt) c[mt][nt] = zero;

  const int ar  = tid >> 2;             // A stage: 2 rows/thread (ar, ar+64)
  const int ac  = (tid & 3) * 8;
  const int bkr = tid >> 3;             // B stage: row k, 16 cols/thread
  const int bns = (tid & 7) * 16;

  const int NT = K >> 5;

  // ---- prologue: stage tile 0 into buffer 0 ----
  async_copy_b128(lds_addr_of(As + ar * 40 + ac),
                  A + (size_t)(m0 + ar) * K + ac);
  async_copy_b128(lds_addr_of(As + (ar + 64) * 40 + ac),
                  A + (size_t)(m0 + ar + 64) * K + ac);
  {
    const v4u* gb = (const v4u*)(Bw + (size_t)bkr * N + n0 + bns);
    v4u b0 = gb[0], b1 = gb[1];
    unsigned short tmp[16];
    *(v4u*)(tmp)     = b0;
    *(v4u*)(tmp + 8) = b1;
#pragma unroll
    for (int j = 0; j < 16; ++j) Bt[(bns + j) * 40 + bkr] = tmp[j];
  }
  wait_asynccnt0();
  __syncthreads();

  for (int it = 0; it < NT; ++it) {
    const int k0 = it << 5;
    const unsigned short* Asc = As + (it & 1) * ABUF;
    const unsigned short* Btc = Bt + (it & 1) * ABUF;
    unsigned short* Asn = As + ((it + 1) & 1) * ABUF;
    unsigned short* Btn = Bt + ((it + 1) & 1) * ABUF;
    const bool more = (it + 1 < NT);

    v4u b0, b1;
    if (more) {
      // async A copy for next tile (no VGPR round-trip, ASYNCcnt)
      async_copy_b128(lds_addr_of(Asn + ar * 40 + ac),
                      A + (size_t)(m0 + ar) * K + k0 + 32 + ac);
      async_copy_b128(lds_addr_of(Asn + (ar + 64) * 40 + ac),
                      A + (size_t)(m0 + ar + 64) * K + k0 + 32 + ac);
      // B rows for next tile into registers (LOADcnt); consumed after compute
      const v4u* gb = (const v4u*)(Bw + (size_t)(k0 + 32 + bkr) * N + n0 + bns);
      b0 = gb[0];
      b1 = gb[1];
      if (it + 2 < NT)  // warm L2 for the tile after next
        __builtin_prefetch(Bw + (size_t)(k0 + 64 + bkr) * N + n0 + bns, 0, 1);
    }

    // ---- compute current tile: 8 x v_wmma_f32_16x16x32_bf16 ----
    Frag a[2], b[4];
#pragma unroll
    for (int mt = 0; mt < 2; ++mt)
      a[mt] = loadA16(Asc + (mBase + mt * 16 + ml) * 40 + half * 16);
#pragma unroll
    for (int nt = 0; nt < 4; ++nt)
      b[nt] = loadB16(Btc + (nBase + nt * 16 + ml) * 40 + half * 16);
#pragma unroll
    for (int mt = 0; mt < 2; ++mt)
#pragma unroll
      for (int nt = 0; nt < 4; ++nt)
        c[mt][nt] = wmma_bf16(a[mt].bf, b[nt].bf, c[mt][nt]);

    if (more) {
      // transpose-store next B tile (waits LOADcnt here, after the WMMAs)
      unsigned short tmp[16];
      *(v4u*)(tmp)     = b0;
      *(v4u*)(tmp + 8) = b1;
#pragma unroll
      for (int j = 0; j < 16; ++j) Btn[(bns + j) * 40 + bkr] = tmp[j];
      wait_asynccnt0();               // next-tile A resident in LDS
    }
    __syncthreads();
  }

  // ---- epilogue ----
#pragma unroll
  for (int mt = 0; mt < 2; ++mt) {
#pragma unroll
    for (int nt = 0; nt < 4; ++nt) {
#pragma unroll
      for (int i = 0; i < 8; ++i) {
        int gm = m0 + mBase + mt * 16 + half * 8 + i;
        int gn = n0 + nBase + nt * 16 + ml;
        float v = c[mt][nt][i];
        if (mode == 0) {
          out32[(size_t)gm * N + gn] = v;
        } else {
          int bb = gm >> 11, l = gm & 2047;
          int h  = gn >> 7,  d = gn & 127;
          size_t idx = (((size_t)bb * NNH + h) * NL + l) * 128 + d;
          if (mode == 1) {
            out16[idx] = f2bf(fmaxf(v, 0.f) * scale);
          } else if (mode == 2) {
            out16[idx] = f2bf(1.f / (1.f + __expf(-v)));
            out32[idx] = -(fmaxf(v, 0.f) + log1pf(__expf(-fabsf(v))));
          } else {  // mode 3
            out16[idx] = f2bf(v);
          }
        }
      }
    }
  }
}

// ---------------------------------------------------------------------------
// Chunked GLA scan: one workgroup per (b,h); chunk C=64.
//   G = cumsum(g) over chunk; qg=q*e^G (q pre-scaled); kg=k*e^-G;
//   P = tril(qg @ kg^T);  o = P @ v + qg @ S;
//   S = S * e^{G_last} + (k*e^{G_last-G})^T @ v
// State kept f32 in LDS with a bf16 mirror (transposed [dv][dk]) for WMMA.
// ---------------------------------------------------------------------------
#define QGS 136   // row stride of [64][128] bf16 tiles
#define TKS 72    // row stride of [128][64] bf16 tiles
#define GSS 132   // row stride of [64][128] f32 gate tile
#define SFS 132   // row stride of [128][128] f32 state

#define SCAN_LDS_BYTES ((64*QGS*2 /*qgs,kgs*/ + 128*TKS /*vst*/ + \
                         128*TKS /*kdt*/ + 64*TKS /*Ps*/ + 128*QGS /*St*/) * 2 + \
                        (64*GSS + 128*SFS) * 4)

__global__ __launch_bounds__(256) void gla_scan_wmma(
    const unsigned short* __restrict__ qg, const unsigned short* __restrict__ kg,
    const unsigned short* __restrict__ vg, const float* __restrict__ gg,
    float* __restrict__ og) {
  extern __shared__ char smem[];
  unsigned short* qgs = (unsigned short*)smem;      // [64][QGS]
  unsigned short* kgs = qgs + 64 * QGS;             // [64][QGS]
  unsigned short* vst = kgs + 64 * QGS;             // [128][TKS]  v^T  [dv][t]
  unsigned short* kdt = vst + 128 * TKS;            // [128][TKS]  kd^T [dk][t]
  unsigned short* Ps  = kdt + 128 * TKS;            // [64][TKS]
  unsigned short* St  = Ps + 64 * TKS;              // [128][QGS]  S^T bf16 [dv][dk]
  float* gs = (float*)(St + 128 * QGS);             // [64][GSS]
  float* Sf = gs + 64 * GSS;                        // [128][SFS]  S f32 [dv][dk]

  const int tid  = threadIdx.x;
  const int lane = tid & 31;
  const int w    = tid >> 5;
  const int ml   = lane & 15;
  const int half = lane >> 4;
  const size_t bh = blockIdx.x;

  // zero the state
  for (int i = tid; i < 128 * SFS; i += 256) Sf[i] = 0.f;
  for (int i = tid; i < 128 * QGS; i += 256) St[i] = 0;
  __syncthreads();

  v8f zero = {0.f, 0.f, 0.f, 0.f, 0.f, 0.f, 0.f, 0.f};

  for (int c0 = 0; c0 < NL; c0 += 64) {
    const size_t base = (bh * NL + c0) * 128;

    // ---- load gates + v (transposed), 8 contiguous elements per slot ----
    for (int i = tid; i < 64 * 16; i += 256) {
      int t  = i >> 4;
      int d8 = (i & 15) * 8;
      const float* gp = gg + base + (size_t)t * 128 + d8;
      *(v4f*)(gs + t * GSS + d8)     = *(const v4f*)gp;
      *(v4f*)(gs + t * GSS + d8 + 4) = *(const v4f*)(gp + 4);
      v4u vv = *(const v4u*)(vg + base + (size_t)t * 128 + d8);
      unsigned short tv[8];
      *(v4u*)tv = vv;
#pragma unroll
      for (int j = 0; j < 8; ++j) vst[(d8 + j) * TKS + t] = tv[j];
    }
    __syncthreads();
    // ---- inclusive cumsum of g along time, per feature column ----
    if (tid < 128) {
      float acc = 0.f;
      for (int t = 0; t < 64; ++t) {
        acc += gs[t * GSS + tid];
        gs[t * GSS + tid] = acc;
      }
    }
    __syncthreads();
    // ---- gated operands ----
    for (int i = tid; i < 64 * 16; i += 256) {
      int t  = i >> 4;
      int d8 = (i & 15) * 8;
      v4u qv8 = *(const v4u*)(qg + base + (size_t)t * 128 + d8);
      v4u kv8 = *(const v4u*)(kg + base + (size_t)t * 128 + d8);
      unsigned short tq[8], tk[8], oq[8], ok[8];
      *(v4u*)tq = qv8;
      *(v4u*)tk = kv8;
#pragma unroll
      for (int j = 0; j < 8; ++j) {
        int d = d8 + j;
        float G  = gs[t * GSS + d];
        float Gl = gs[63 * GSS + d];
        float qv = bf2f(tq[j]);
        float kv = bf2f(tk[j]);
        oq[j] = f2bf(qv * __expf(G));
        ok[j] = f2bf(kv * __expf(-G));
        kdt[d * TKS + t] = f2bf(kv * __expf(Gl - G));
      }
      *(v4u*)(qgs + t * QGS + d8) = *(v4u*)oq;
      *(v4u*)(kgs + t * QGS + d8) = *(v4u*)ok;
    }
    __syncthreads();

    // ---- P = tril(qg @ kg^T)  [64x64], 16 tiles over 8 waves ----
    {
      const int pm = w >> 1;
      v8f pc[2];
      pc[0] = zero; pc[1] = zero;
#pragma unroll
      for (int kk = 0; kk < 128; kk += 32) {
        Frag a = loadA16(qgs + (pm * 16 + ml) * QGS + kk + half * 16);
#pragma unroll
        for (int j = 0; j < 2; ++j) {
          int pn = (w & 1) * 2 + j;
          Frag b = loadB16(kgs + (pn * 16 + ml) * QGS + kk + half * 16);
          pc[j] = wmma_bf16(a.bf, b.bf, pc[j]);
        }
      }
#pragma unroll
      for (int j = 0; j < 2; ++j)
#pragma unroll
        for (int i = 0; i < 8; ++i) {
          int tr = pm * 16 + half * 8 + i;
          int tc = ((w & 1) * 2 + j) * 16 + ml;
          Ps[tr * TKS + tc] = f2bf(tr >= tc ? pc[j][i] : 0.f);
        }
    }
    __syncthreads();

    // ---- o = P @ v + qg @ S   [64x128], 32 tiles over 8 waves ----
    {
      const int om = w & 3;
      const int ng = w >> 2;
      v8f oc[4];
#pragma unroll
      for (int j = 0; j < 4; ++j) oc[j] = zero;
      // intra-chunk: K = 64 (time)
#pragma unroll
      for (int kk = 0; kk < 64; kk += 32) {
        Frag a = loadA16(Ps + (om * 16 + ml) * TKS + kk + half * 16);
#pragma unroll
        for (int j = 0; j < 4; ++j) {
          int dvb = (ng * 4 + j) * 16;
          Frag b = loadB16(vst + (dvb + ml) * TKS + kk + half * 16);
          oc[j] = wmma_bf16(a.bf, b.bf, oc[j]);
        }
      }
      // inter-chunk: K = 128 (dk), B = S^T rows
#pragma unroll
      for (int kk = 0; kk < 128; kk += 32) {
        Frag a = loadA16(qgs + (om * 16 + ml) * QGS + kk + half * 16);
#pragma unroll
        for (int j = 0; j < 4; ++j) {
          int dvb = (ng * 4 + j) * 16;
          Frag b = loadB16(St + (dvb + ml) * QGS + kk + half * 16);
          oc[j] = wmma_bf16(a.bf, b.bf, oc[j]);
        }
      }
#pragma unroll
      for (int j = 0; j < 4; ++j)
#pragma unroll
        for (int i = 0; i < 8; ++i) {
          int t  = om * 16 + half * 8 + i;
          int dv = (ng * 4 + j) * 16 + ml;
          og[base + (size_t)t * 128 + dv] = oc[j][i];
        }
    }

    // ---- U = kd^T @ v  [128x128], 64 tiles over 8 waves ----
    {
      v8f uc[8];
#pragma unroll
      for (int j = 0; j < 8; ++j) uc[j] = zero;
#pragma unroll
      for (int kk = 0; kk < 64; kk += 32) {
        Frag a = loadA16(kdt + (w * 16 + ml) * TKS + kk + half * 16);
#pragma unroll
        for (int j = 0; j < 8; ++j) {
          Frag b = loadB16(vst + (j * 16 + ml) * TKS + kk + half * 16);
          uc[j] = wmma_bf16(a.bf, b.bf, uc[j]);
        }
      }
      __syncthreads();  // all waves done reading St in the o-phase
      // S = S * e^{G_last} + U   (stored transposed [dv][dk])
#pragma unroll
      for (int j = 0; j < 8; ++j)
#pragma unroll
        for (int i = 0; i < 8; ++i) {
          int dk = w * 16 + half * 8 + i;
          int dv = j * 16 + ml;
          float decay = __expf(gs[63 * GSS + dk]);
          float s = Sf[dv * SFS + dk] * decay + uc[j][i];
          Sf[dv * SFS + dk] = s;
          St[dv * QGS + dk] = f2bf(s);
        }
    }
    __syncthreads();  // state + gs stable before next chunk's loads
  }
}

// ---------------------------------------------------------------------------
// RMSNorm over HID with head-layout gather; emits bf16 for the final GEMM.
// ---------------------------------------------------------------------------
__global__ __launch_bounds__(256) void rmsnorm_bf16(
    const float* __restrict__ o, const float* __restrict__ gw,
    unsigned short* __restrict__ y) {
  const int row = blockIdx.x;          // b*L + l
  const int b = row >> 11, l = row & 2047;
  __shared__ float red[256];
  float vals[8];
  float ss = 0.f;
#pragma unroll
  for (int j = 0; j < 8; ++j) {
    int e = threadIdx.x + j * 256;
    int h = e >> 7, d = e & 127;
    float v = o[(((size_t)b * NNH + h) * NL + l) * 128 + d];
    vals[j] = v;
    ss += v * v;
  }
  red[threadIdx.x] = ss;
  __syncthreads();
  for (int s = 128; s > 0; s >>= 1) {
    if (threadIdx.x < s) red[threadIdx.x] += red[threadIdx.x + s];
    __syncthreads();
  }
  float scale = rsqrtf(red[0] / (float)NHID + REPS);
#pragma unroll
  for (int j = 0; j < 8; ++j) {
    int e = threadIdx.x + j * 256;
    y[(size_t)row * NHID + e] = f2bf(vals[j] * scale * gw[e]);
  }
}

// ---------------------------------------------------------------------------
// Host entry
// ---------------------------------------------------------------------------
extern "C" void kernel_launch(void* const* d_in, const int* in_sizes, int n_in,
                              void* d_out, int out_size, void* d_ws, size_t ws_size,
                              hipStream_t stream) {
  (void)in_sizes; (void)n_in; (void)out_size; (void)ws_size;
  const float* h_f32  = (const float*)d_in[0];   // [B,L,HID]
  const float* Wq_f32 = (const float*)d_in[1];
  const float* Wk_f32 = (const float*)d_in[2];
  const float* Wi_f32 = (const float*)d_in[3];
  const float* Wo_f32 = (const float*)d_in[4];
  const float* gwt    = (const float*)d_in[5];
  float* out = (float*)d_out;

  const int MROWS = NB * NL;                 // 4096
  const size_t NH_ELEM = (size_t)MROWS * NHID;          // 8,388,608
  const size_t NW_ELEM = (size_t)NHID * NHID;           // 4,194,304
  const size_t NBHLD   = (size_t)NB * NNH * NL * NDK;   // 8,388,608

  size_t off = 0;
  auto carve = [&](size_t bytes) -> void* {
    void* p = (char*)d_ws + off;
    off += (bytes + 255) & ~(size_t)255;
    return p;
  };
  unsigned short* hbf  = (unsigned short*)carve(NH_ELEM * 2);
  unsigned short* wqb  = (unsigned short*)carve(NW_ELEM * 2);
  unsigned short* wkb  = (unsigned short*)carve(NW_ELEM * 2);
  unsigned short* wib  = (unsigned short*)carve(NW_ELEM * 2);
  unsigned short* wob  = (unsigned short*)carve(NW_ELEM * 2);
  unsigned short* qb   = (unsigned short*)carve(NBHLD * 2);
  unsigned short* kb   = (unsigned short*)carve(NBHLD * 2);
  unsigned short* vb   = (unsigned short*)carve(NBHLD * 2);
  float*          gbuf = (float*)carve(NBHLD * 4);
  float*          obuf = (float*)carve(NBHLD * 4);
  unsigned short* ybf  = (unsigned short*)carve(NH_ELEM * 2);

  // 1) bf16 conversion of activations + weights
  cvt_f32_bf16<<<(int)((NH_ELEM + 255) / 256), 256, 0, stream>>>(h_f32, hbf, (int)NH_ELEM);
  cvt_f32_bf16<<<(int)((NW_ELEM + 255) / 256), 256, 0, stream>>>(Wq_f32, wqb, (int)NW_ELEM);
  cvt_f32_bf16<<<(int)((NW_ELEM + 255) / 256), 256, 0, stream>>>(Wk_f32, wkb, (int)NW_ELEM);
  cvt_f32_bf16<<<(int)((NW_ELEM + 255) / 256), 256, 0, stream>>>(Wi_f32, wib, (int)NW_ELEM);
  cvt_f32_bf16<<<(int)((NW_ELEM + 255) / 256), 256, 0, stream>>>(Wo_f32, wob, (int)NW_ELEM);

  // 2) fused projections (WMMA GEMM + epilogue into [B,NH,L,D] layout)
  dim3 ggrid(NHID / 128, MROWS / 128);   // 16 x 32
  gemm_bf16_wmma<<<ggrid, 256, 0, stream>>>(hbf, wqb, MROWS, NHID, NHID, 1, QSCALE, qb, nullptr);
  gemm_bf16_wmma<<<ggrid, 256, 0, stream>>>(hbf, wkb, MROWS, NHID, NHID, 2, 1.f, kb, gbuf);
  gemm_bf16_wmma<<<ggrid, 256, 0, stream>>>(hbf, wib, MROWS, NHID, NHID, 3, 1.f, vb, nullptr);

  // 3) chunked GLA scan (one WG per (b,h); ~212KB LDS of the 320KB WGP pool)
  hipFuncSetAttribute((const void*)gla_scan_wmma,
                      hipFuncAttributeMaxDynamicSharedMemorySize, SCAN_LDS_BYTES);
  gla_scan_wmma<<<NB * NNH, 256, SCAN_LDS_BYTES, stream>>>(qb, kb, vb, gbuf, obuf);

  // 4) RMSNorm (+ affine) into bf16
  rmsnorm_bf16<<<MROWS, 256, 0, stream>>>(obuf, gwt, ybf);

  // 5) output projection, f32 row-major result
  gemm_bf16_wmma<<<ggrid, 256, 0, stream>>>(ybf, wob, MROWS, NHID, NHID, 0, 1.f, nullptr, out);
}
